// T5Attention_67929202754011
// MI455X (gfx1250) — compile-verified, matrix operand order checked by reading, not run on hardware
//
#include <hip/hip_runtime.h>
#include <hip/hip_fp16.h>
#include <math.h>

// ---------------------------------------------------------------------------
// T5 encoder self-attention for MI455X (gfx1250, wave32, WMMA).
// f32->f16 convert (weights transposed) -> 3x double-buffered async WMMA GEMM
// (V written head-transposed) -> flash attention (S^T form, async K/V staging,
// LDS-resident bias, register P->B-frag) -> double-buffered WMMA GEMM out.
// ---------------------------------------------------------------------------

typedef __attribute__((ext_vector_type(16))) _Float16 v16h;
typedef __attribute__((ext_vector_type(8)))  _Float16 h8;
typedef __attribute__((ext_vector_type(8)))  float    v8f;
typedef __attribute__((ext_vector_type(4)))  float    f4;

union Frag16 { v16h v; h8 h[2]; };

__device__ __forceinline__ v8f wmma16x16x32(const Frag16& a, const Frag16& b, v8f c) {
  return __builtin_amdgcn_wmma_f32_16x16x32_f16(false, a.v, false, b.v, (short)0, c,
                                                false, false);
}

// CDNA5 async global->LDS copy (bypasses VGPRs, tracked by ASYNCcnt).
__device__ __forceinline__ void async_load_b128(const void* g, void* lds) {
  unsigned int l = (unsigned int)(unsigned long long)lds;
  asm volatile("global_load_async_to_lds_b128 %0, %1, off"
               :: "v"(l), "v"(g) : "memory");
}
__device__ __forceinline__ void async_wait0() {
  asm volatile("s_wait_asynccnt 0x0" ::: "memory");
}

constexpr int B_  = 2;
constexpr int S_  = 2048;
constexpr int D_  = 1024;
constexpr int H_  = 16;
constexpr int DH_ = 64;
constexpr int HD_ = H_ * DH_;    // 1024
constexpr int M_  = B_ * S_;     // 4096
constexpr int RELN = 2 * S_ - 1; // 4095
constexpr int BIASL = 2176;      // 2175 floats cover a block's rel range

// ---------------------------------------------------------------------------
// 1a) f32 -> f16 conversion (row-major kept), 8 elems/thread
// ---------------------------------------------------------------------------
__global__ void __launch_bounds__(256) cvt_f32_f16(const float* __restrict__ src,
                                                   _Float16* __restrict__ dst, int n) {
  int i = (blockIdx.x * 256 + threadIdx.x) * 8;
  if (i + 8 > n) return;
  f4 a = *(const f4*)(src + i);
  f4 b = *(const f4*)(src + i + 4);
  h8 o;
  o[0] = (_Float16)a[0]; o[1] = (_Float16)a[1];
  o[2] = (_Float16)a[2]; o[3] = (_Float16)a[3];
  o[4] = (_Float16)b[0]; o[5] = (_Float16)b[1];
  o[6] = (_Float16)b[2]; o[7] = (_Float16)b[3];
  *(h8*)(dst + i) = o;
}

// ---------------------------------------------------------------------------
// 1b) f32[1024][1024] -> transposed f16[1024][1024] (weights, one-time)
// ---------------------------------------------------------------------------
__global__ void __launch_bounds__(256) cvt_f32_f16_T(const float* __restrict__ src,
                                                     _Float16* __restrict__ dst) {
  __shared__ float tile[32][33];
  int k0 = blockIdx.y * 32, n0 = blockIdx.x * 32;
  int tx = threadIdx.x & 31, ty = threadIdx.x >> 5;   // 32 x 8
#pragma unroll
  for (int i = 0; i < 32; i += 8)
    tile[ty + i][tx] = src[(size_t)(k0 + ty + i) * D_ + n0 + tx];
  __syncthreads();
#pragma unroll
  for (int i = 0; i < 32; i += 8)
    dst[(size_t)(n0 + ty + i) * D_ + k0 + tx] = (_Float16)tile[tx][ty + i];
}

// ---------------------------------------------------------------------------
// 2) Expand rel_bias_table[32][H] -> biasE[H][4095] indexed by (rel + S-1)
// ---------------------------------------------------------------------------
__global__ void __launch_bounds__(256) bias_expand(const float* __restrict__ table,
                                                   float* __restrict__ biasE) {
  int tid = blockIdx.x * 256 + threadIdx.x;
  if (tid >= H_ * RELN) return;
  int h = tid / RELN, r = tid % RELN;
  int rel = r - (S_ - 1);
  const int nb = 16, me = 8;
  int rb = (rel > 0) ? nb : 0;
  int rp = rel < 0 ? -rel : rel;
  int b;
  if (rp < me) {
    b = rp;
  } else {
    float t = __logf((float)rp * (1.0f / me)) * (1.0f / __logf(16.0f)) * (float)(nb - me);
    b = me + (int)t;
    if (b > nb - 1) b = nb - 1;
  }
  biasE[h * RELN + r] = table[(rb + b) * H_ + h];
}

// ---------------------------------------------------------------------------
// 3) WMMA GEMM: C[M,N] = A[M,K](f16) * BT[N,K](f16, pre-transposed).
//    128x128x64 tile, 8 waves (4x2), wave = 32x64. Double-buffered async
//    global->LDS staging for both operands (no VGPR bounce, no scalar stores).
//    VOUT=true scatters the result as VT[b][h][dh][s] for the attention pass.
// ---------------------------------------------------------------------------
template <typename OutT, bool VOUT>
__global__ void __launch_bounds__(256)
gemm_wmma(const _Float16* __restrict__ A, const _Float16* __restrict__ BT,
          OutT* __restrict__ C, int Mdim, int Ndim, int Kdim) {
  __shared__ __align__(16) _Float16 As[2][128][64];
  __shared__ __align__(16) _Float16 Bs[2][128][64];   // [n][k]
  const int t    = threadIdx.x;
  const int lane = t & 31, w = t >> 5;
  const int wm   = w & 3, wn = w >> 2;
  const int m0   = blockIdx.y * 128, n0 = blockIdx.x * 128;
  const int lr   = lane & 15, ko = (lane >> 4) * 8;
  const int srow = t >> 1, soff = (t & 1) * 32;       // staging coords

  auto stage = [&](int k0, int buf) {
    const _Float16* sa = A  + (size_t)(m0 + srow) * Kdim + k0 + soff;
    const _Float16* sb = BT + (size_t)(n0 + srow) * Kdim + k0 + soff;
#pragma unroll
    for (int c = 0; c < 32; c += 8) {
      async_load_b128(sa + c, &As[buf][srow][soff + c]);
      async_load_b128(sb + c, &Bs[buf][srow][soff + c]);
    }
  };

  v8f acc[2][4] = {};
  stage(0, 0);
  async_wait0();
  __syncthreads();

  const int T = Kdim / 64;
  for (int i = 0; i < T; ++i) {
    int buf = i & 1;
    if (i + 1 < T) stage((i + 1) * 64, buf ^ 1);   // overlap with compute
#pragma unroll
    for (int ks = 0; ks < 64; ks += 32) {
      Frag16 a[2], b[4];
#pragma unroll
      for (int mt = 0; mt < 2; ++mt) {
        int row = wm * 32 + mt * 16 + lr;
        a[mt].h[0] = *(const h8*)&As[buf][row][ks + ko];
        a[mt].h[1] = *(const h8*)&As[buf][row][ks + ko + 16];
      }
#pragma unroll
      for (int nt = 0; nt < 4; ++nt) {
        int col = wn * 64 + nt * 16 + lr;
        b[nt].h[0] = *(const h8*)&Bs[buf][col][ks + ko];
        b[nt].h[1] = *(const h8*)&Bs[buf][col][ks + ko + 16];
      }
#pragma unroll
      for (int mt = 0; mt < 2; ++mt)
#pragma unroll
        for (int nt = 0; nt < 4; ++nt)
          acc[mt][nt] = wmma16x16x32(a[mt], b[nt], acc[mt][nt]);
    }
    async_wait0();
    __syncthreads();
  }

#pragma unroll
  for (int mt = 0; mt < 2; ++mt)
#pragma unroll
    for (int nt = 0; nt < 4; ++nt)
#pragma unroll
      for (int j = 0; j < 8; ++j) {
        int m = m0 + wm * 32 + mt * 16 + j + (lane >> 4) * 8;
        int n = n0 + wn * 64 + nt * 16 + lr;
        if constexpr (VOUT) {
          int bb = m >> 11, s = m & (S_ - 1);       // token -> (batch, seq)
          int hh = n >> 6,  dh = n & (DH_ - 1);     // col   -> (head, dh)
          C[((size_t)((bb * H_ + hh) * DH_ + dh)) * S_ + s] = (OutT)acc[mt][nt][j];
        } else {
          C[(size_t)m * Ndim + n] = (OutT)acc[mt][nt][j];
        }
      }
}

// ---------------------------------------------------------------------------
// 4) Flash attention in S^T form, double-buffered async K/V staging.
//    S^T = K * Q^T  ->  lane = query col, j = key row; per-query scalar state;
//    P^T converts in registers to the B-fragment of O^T = V^T * P^T.
//    Bias row segment staged once into LDS (block-invariant indexing).
// ---------------------------------------------------------------------------
__global__ void __launch_bounds__(256)
flash_attn(const _Float16* __restrict__ Q, const _Float16* __restrict__ K,
           const _Float16* __restrict__ VT, const float* __restrict__ biasE,
           _Float16* __restrict__ Ctx) {
  __shared__ __align__(16) _Float16 Ks[2][32][64];   // [key][dh]
  __shared__ __align__(16) _Float16 Vs[2][64][32];   // [dh][key] (from VT)
  __shared__ float biasL[BIASL];

  const int t = threadIdx.x, lane = t & 31, w = t >> 5;
  const int lr = lane & 15, ko = (lane >> 4) * 8;
  const int b = blockIdx.z, h = blockIdx.y;
  const int q0 = blockIdx.x * 128, qrow = q0 + w * 16;

  // stage this block's bias segment: biasL[i] = bias(rel = i - (q0+127))
  {
    const float* biasH = biasE + h * RELN + (S_ - 1);
    for (int i = t; i < BIASL - 1; i += 256) biasL[i] = biasH[i - (q0 + 127)];
  }

  const _Float16* Kb  = K  + (size_t)(b * S_) * HD_ + h * DH_;
  const _Float16* VTb = VT + (size_t)((b * H_ + h) * DH_) * S_;

  auto stageKV = [&](int kb, int buf) {
    int krow = t >> 3, koff8 = (t & 7) * 8;          // 32 x 64 K tile
    async_load_b128(Kb + (size_t)(kb + krow) * HD_ + koff8, &Ks[buf][krow][koff8]);
    int drow = t >> 2, soff8 = (t & 3) * 8;          // 64 x 32 V^T tile
    async_load_b128(VTb + (size_t)drow * S_ + kb + soff8, &Vs[buf][drow][soff8]);
  };

  // Q^T B-fragments (lane = query, regs = dh), loaded once from global
  Frag16 qf[2];
  {
    const _Float16* qp = Q + ((size_t)(b * S_ + qrow + lr) * HD_) + h * DH_;
    qf[0].h[0] = *(const h8*)(qp + ko);
    qf[0].h[1] = *(const h8*)(qp + ko + 16);
    qf[1].h[0] = *(const h8*)(qp + 32 + ko);
    qf[1].h[1] = *(const h8*)(qp + 32 + ko + 16);
  }

  v8f o[4] = {};
  float mx = -INFINITY, lsum = 0.0f;

  stageKV(0, 0);
  async_wait0();
  __syncthreads();

  for (int kb = 0; kb < S_; kb += 32) {
    int buf = (kb >> 5) & 1;
    if (kb + 32 < S_) stageKV(kb + 32, buf ^ 1);     // overlap with compute

    // S^T tiles: rows = keys (tk*16..), cols = queries
    v8f sc[2];
#pragma unroll
    for (int tk = 0; tk < 2; ++tk) {
      int key = tk * 16 + lr;
      Frag16 ka0, ka1;
      ka0.h[0] = *(const h8*)&Ks[buf][key][ko];
      ka0.h[1] = *(const h8*)&Ks[buf][key][ko + 16];
      ka1.h[0] = *(const h8*)&Ks[buf][key][32 + ko];
      ka1.h[1] = *(const h8*)&Ks[buf][key][32 + ko + 16];
      v8f c = {};
      c = wmma16x16x32(ka0, qf[0], c);
      c = wmma16x16x32(ka1, qf[1], c);
      sc[tk] = c;
    }

    // scale + bias from LDS (index independent of blockIdx)
#pragma unroll
    for (int tk = 0; tk < 2; ++tk) {
      int bbase = kb + tk * 16 + ((lane >> 4) << 3) - (w * 16 + lr) + 127;
#pragma unroll
      for (int j = 0; j < 8; ++j)
        sc[tk][j] = sc[tk][j] * 0.125f + biasL[bbase + j];
    }

    // online softmax, per-query scalar state (lane l and l^16 share a query)
    float mloc = sc[0][0];
#pragma unroll
    for (int j = 1; j < 8; ++j) mloc = fmaxf(mloc, sc[0][j]);
#pragma unroll
    for (int j = 0; j < 8; ++j) mloc = fmaxf(mloc, sc[1][j]);
    mloc = fmaxf(mloc, __shfl_xor(mloc, 16, 32));
    float nm    = fmaxf(mx, mloc);
    float alpha = __expf(mx - nm);
    mx = nm;

    Frag16 pf;
    float rs = 0.0f;
#pragma unroll
    for (int j = 0; j < 8; ++j) {
      float p0 = __expf(sc[0][j] - nm);
      float p1 = __expf(sc[1][j] - nm);
      rs += p0 + p1;
      pf.h[0][j] = (_Float16)p0;
      pf.h[1][j] = (_Float16)p1;
    }
    rs += __shfl_xor(rs, 16, 32);
    lsum = lsum * alpha + rs;

#pragma unroll
    for (int tdh = 0; tdh < 4; ++tdh)
#pragma unroll
      for (int j = 0; j < 8; ++j) o[tdh][j] *= alpha;

    // O^T += V^T(16dh x 32k) * P^T(32k x 16q)
#pragma unroll
    for (int tdh = 0; tdh < 4; ++tdh) {
      int dhrow = tdh * 16 + lr;
      Frag16 va;
      va.h[0] = *(const h8*)&Vs[buf][dhrow][ko];
      va.h[1] = *(const h8*)&Vs[buf][dhrow][ko + 16];
      o[tdh] = wmma16x16x32(va, pf, o[tdh]);
    }
    async_wait0();
    __syncthreads();
  }

  // normalize + store context row-major [token][h*64+dh]
  float inv = 1.0f / lsum;
#pragma unroll
  for (int tdh = 0; tdh < 4; ++tdh)
#pragma unroll
    for (int j = 0; j < 8; ++j) {
      int dhcol = tdh * 16 + j + (lane >> 4) * 8;
      Ctx[((size_t)(b * S_ + qrow + lr) * HD_) + h * DH_ + dhcol] =
          (_Float16)(o[tdh][j] * inv);
    }
}

// ---------------------------------------------------------------------------
// Launcher
// ---------------------------------------------------------------------------
extern "C" void kernel_launch(void* const* d_in, const int* in_sizes, int n_in,
                              void* d_out, int out_size, void* d_ws, size_t ws_size,
                              hipStream_t stream) {
  (void)in_sizes; (void)n_in; (void)out_size; (void)ws_size;
  const float* hs    = (const float*)d_in[0];
  const float* Wq    = (const float*)d_in[1];
  const float* Wk    = (const float*)d_in[2];
  const float* Wv    = (const float*)d_in[3];
  const float* Wo    = (const float*)d_in[4];
  const float* table = (const float*)d_in[5];

  char*  ws  = (char*)d_ws;
  size_t off = 0;
  auto alloc = [&](size_t bytes) -> void* {
    void* p = ws + off;
    off += (bytes + 255) & ~(size_t)255;
    return p;
  };

  _Float16* hs16   = (_Float16*)alloc((size_t)M_ * D_ * 2);
  _Float16* wq16T  = (_Float16*)alloc((size_t)D_ * HD_ * 2);
  _Float16* wk16T  = (_Float16*)alloc((size_t)D_ * HD_ * 2);
  _Float16* wv16T  = (_Float16*)alloc((size_t)D_ * HD_ * 2);
  _Float16* wo16T  = (_Float16*)alloc((size_t)HD_ * D_ * 2);
  _Float16* q16    = (_Float16*)alloc((size_t)M_ * HD_ * 2);
  _Float16* k16    = (_Float16*)alloc((size_t)M_ * HD_ * 2);
  _Float16* vT16   = (_Float16*)alloc((size_t)M_ * HD_ * 2);
  _Float16* ctx16  = (_Float16*)alloc((size_t)M_ * HD_ * 2);
  float*    biasE  = (float*)alloc((size_t)H_ * RELN * 4);

  const int nHS = M_ * D_;
  cvt_f32_f16<<<nHS / 8 / 256, 256, 0, stream>>>(hs, hs16, nHS);
  dim3 gT(D_ / 32, D_ / 32);
  cvt_f32_f16_T<<<gT, 256, 0, stream>>>(Wq, wq16T);
  cvt_f32_f16_T<<<gT, 256, 0, stream>>>(Wk, wk16T);
  cvt_f32_f16_T<<<gT, 256, 0, stream>>>(Wv, wv16T);
  cvt_f32_f16_T<<<gT, 256, 0, stream>>>(Wo, wo16T);
  bias_expand<<<(H_ * RELN + 255) / 256, 256, 0, stream>>>(table, biasE);

  dim3 gGemm(HD_ / 128, M_ / 128);
  gemm_wmma<_Float16, false><<<gGemm, 256, 0, stream>>>(hs16, wq16T, q16, M_, HD_, D_);
  gemm_wmma<_Float16, false><<<gGemm, 256, 0, stream>>>(hs16, wk16T, k16, M_, HD_, D_);
  gemm_wmma<_Float16, true ><<<gGemm, 256, 0, stream>>>(hs16, wv16T, vT16, M_, HD_, D_);

  dim3 gAttn(S_ / 128, H_, B_);
  flash_attn<<<gAttn, 256, 0, stream>>>(q16, k16, vT16, biasE, ctx16);

  dim3 gOut(D_ / 128, M_ / 128);
  gemm_wmma<float, false><<<gOut, 256, 0, stream>>>(ctx16, wo16T, (float*)d_out, M_, D_, HD_);
}
